// GATLayer_76141180223896
// MI455X (gfx1250) — compile-verified, matrix operand order checked by reading, not run on hardware
//
#include <hip/hip_runtime.h>
#include <hip/hip_bf16.h>

typedef _Float16 v16h __attribute__((ext_vector_type(16)));
typedef _Float16 v8h  __attribute__((ext_vector_type(8)));
typedef float    v8f  __attribute__((ext_vector_type(8)));
typedef float    v4f  __attribute__((ext_vector_type(4)));

#define IN_F   256
#define HF     256
#define HEADS  8
#define OUT_F  32
#define NEG_SLOPE 0.2f

// ---------------------------------------------------------------------------
// Convert Wv [256 (k) x 256 (n)] f32 row-major -> WT [n][k] f16 so that the
// WMMA B fragment is two contiguous 16B loads per lane.
// ---------------------------------------------------------------------------
__global__ void gat_wt_convert(const float* __restrict__ Wv,
                               _Float16* __restrict__ WT) {
  int idx = blockIdx.x * blockDim.x + threadIdx.x;   // idx = n*256 + k
  if (idx < IN_F * HF) {
    int n = idx >> 8, k = idx & 255;
    WT[idx] = (_Float16)Wv[k * HF + n];
  }
}

// ---------------------------------------------------------------------------
// H(f16) = X @ Wv + bv  via v_wmma_f32_16x16x32_f16.
// ct-outer / K-inner: the wave's whole 16x256 A stripe lives in 64 VGPRs
// (loaded once, non-temporal), one v8f accumulator at a time, all 8 B
// fragments of a column tile issued as one load clause ahead of the 8 WMMAs.
// A fragment (16-bit A 16x32 layout): lane L -> row = L&15,
//   kbase = (L>>4)*8, elems 0..7 -> K=kbase+0..7, elems 8..15 -> K=16+kbase+0..7
// B fragment: lane L -> col n = L&15, same K pattern, loaded from WT[n][k].
// C/D (16x16 f32): VGPR r -> M = r + 8*(L>>4), N = L&15.
// ---------------------------------------------------------------------------
__global__ __launch_bounds__(256) void gat_gemm256(
    const float* __restrict__ X, const _Float16* __restrict__ WT,
    const float* __restrict__ bias, _Float16* __restrict__ H, int nrows) {
  const int lane    = threadIdx.x & 31;
  const int wave    = threadIdx.x >> 5;
  const int rowTile = blockIdx.x * 8 + wave;
  const int nRowTiles = (nrows + 15) >> 4;
  if (rowTile >= nRowTiles) return;           // wave-uniform: EXEC stays all-1s

  const int r0    = rowTile << 4;
  const int m     = lane & 15;
  const int kbase = (lane >> 4) << 3;

  int arow = r0 + m;
  if (arow >= nrows) arow = nrows - 1;        // clamp loads on tail tile
  const float* xrow = X + (size_t)arow * IN_F;

  // Load the whole 256-wide A stripe once (single-use: non-temporal).
  v16h a[8];
#pragma unroll
  for (int t = 0; t < 8; ++t) {
    const int k0 = t * 32;
    v4f a0 = __builtin_nontemporal_load((const v4f*)(xrow + k0 + kbase));
    v4f a1 = __builtin_nontemporal_load((const v4f*)(xrow + k0 + kbase + 4));
    v4f a2 = __builtin_nontemporal_load((const v4f*)(xrow + k0 + kbase + 16));
    v4f a3 = __builtin_nontemporal_load((const v4f*)(xrow + k0 + kbase + 20));
#pragma unroll
    for (int j = 0; j < 4; ++j) {
      a[t][j]      = (_Float16)a0[j];
      a[t][j + 4]  = (_Float16)a1[j];
      a[t][j + 8]  = (_Float16)a2[j];
      a[t][j + 12] = (_Float16)a3[j];
    }
  }

  const int rup = (lane >> 4) << 3;
#pragma unroll
  for (int ct = 0; ct < 16; ++ct) {
    const _Float16* wrow = WT + (size_t)(ct * 16 + m) * IN_F + kbase;
    union { v16h v; v8h h[2]; } b[8];
#pragma unroll
    for (int t = 0; t < 8; ++t) {             // one clause of 16 b128 loads
      b[t].h[0] = *(const v8h*)(wrow + t * 32);
      b[t].h[1] = *(const v8h*)(wrow + t * 32 + 16);
    }
    v8f acc = (v8f)0.0f;
#pragma unroll
    for (int t = 0; t < 8; ++t)
      acc = __builtin_amdgcn_wmma_f32_16x16x32_f16(
          false, a[t], false, b[t].v, (short)0, acc, false, false);

    const int col  = ct * 16 + m;
    const float bc = bias[col];
#pragma unroll
    for (int r = 0; r < 8; ++r) {
      int row = r0 + r + rup;
      if (row < nrows)
        H[(size_t)row * HF + col] = (_Float16)(acc[r] + bc);
    }
  }
}

// ---------------------------------------------------------------------------
// q/k projection from f16 h: out[n,h] = H[n,:] . W[:,h] + b[h].
// One wave per node, W (8KB) staged in LDS; 8 shuffle reductions per node.
// ---------------------------------------------------------------------------
__global__ __launch_bounds__(256) void gat_qk(
    const _Float16* __restrict__ H, const float* __restrict__ W,
    const float* __restrict__ b, float* __restrict__ out, int nnodes) {
  __shared__ float Wl[HF * HEADS];
  __shared__ float bl[HEADS];
  const int t = threadIdx.x;
#pragma unroll
  for (int j = 0; j < 8; ++j) Wl[t + 256 * j] = W[t + 256 * j];
  if (t < HEADS) bl[t] = b[t];
  __syncthreads();

  const int lane = t & 31;
  const int node = blockIdx.x * 8 + (t >> 5);
  if (node >= nnodes) return;

  float hv[8];
  const _Float16* hrow = H + (size_t)node * HF;
#pragma unroll
  for (int j = 0; j < 8; ++j) hv[j] = (float)hrow[lane + 32 * j];

#pragma unroll
  for (int c = 0; c < HEADS; ++c) {
    float p = 0.f;
#pragma unroll
    for (int j = 0; j < 8; ++j) p += hv[j] * Wl[(lane + 32 * j) * HEADS + c];
#pragma unroll
    for (int off = 16; off > 0; off >>= 1) p += __shfl_xor(p, off, 32);
    if (lane == 0) out[(size_t)node * HEADS + c] = p + bl[c];
  }
}

// ---------------------------------------------------------------------------
// CSR row offsets from sorted dst_idx (binary search per dst node).
// ---------------------------------------------------------------------------
__global__ void gat_segstart(const int* __restrict__ dst_idx, int E, int ndst,
                             int* __restrict__ seg) {
  int d = blockIdx.x * blockDim.x + threadIdx.x;
  if (d > ndst) return;
  if (d == ndst) { seg[d] = E; return; }
  int lo = 0, hi = E;
  while (lo < hi) {
    int mid = (lo + hi) >> 1;
    if (dst_idx[mid] < d) lo = mid + 1; else hi = mid;
  }
  seg[d] = lo;
}

// ---------------------------------------------------------------------------
// Per-dst-node fused edge softmax + weighted aggregation + head mean.
// Block of 256 = 8 waves; wave h handles head h, lane f handles feature f.
// Scores are recomputed per pass (3x 1.6M*8 cheap VALU ops) instead of
// materializing E*H attention weights.  h_src rows are gathered in f16
// (512B/edge, L2-resident) to halve the dominant gather traffic.
// ---------------------------------------------------------------------------
__global__ __launch_bounds__(256) void gat_edge_aggr(
    const _Float16* __restrict__ Hsrc, const float* __restrict__ q,
    const float* __restrict__ k, const int* __restrict__ src_idx,
    const int* __restrict__ seg, float* __restrict__ out, int ndst) {
  __shared__ float qd[HEADS], msh[HEADS], ssh[HEADS];
  __shared__ float red[256];

  const int d = blockIdx.x;
  const int t = threadIdx.x;
  const int h = t >> 5;
  const int f = t & 31;
  const int start = seg[d];
  const int deg   = seg[d + 1] - start;

  if (t < HEADS) qd[t] = q[(size_t)d * HEADS + t];
  __syncthreads();
  const float qh = qd[h];

  // pass 1: per-head max of leaky_relu(q+k)
  float mx = -3.0e38f;
  for (int i = f; i < deg; i += 32) {
    int s = src_idx[start + i];
    float v = qh + k[(size_t)s * HEADS + h];
    float e = v > 0.f ? v : NEG_SLOPE * v;
    mx = fmaxf(mx, e);
  }
#pragma unroll
  for (int off = 16; off > 0; off >>= 1) mx = fmaxf(mx, __shfl_xor(mx, off, 32));
  if (f == 0) msh[h] = mx;
  __syncthreads();
  const float mh = msh[h];

  // pass 2: per-head sum of exp(e - m)
  float sm = 0.f;
  for (int i = f; i < deg; i += 32) {
    int s = src_idx[start + i];
    float v = qh + k[(size_t)s * HEADS + h];
    float e = v > 0.f ? v : NEG_SLOPE * v;
    sm += __expf(e - mh);
  }
#pragma unroll
  for (int off = 16; off > 0; off >>= 1) sm += __shfl_xor(sm, off, 32);
  if (f == 0) ssh[h] = sm;
  __syncthreads();
  const float inv = 1.f / (ssh[h] + 1e-9f);

  // pass 3: aggregate attn-weighted h_src rows (coalesced 512B row per edge)
  float acc = 0.f;
  for (int i = 0; i < deg; ++i) {
    int s = src_idx[start + i];
    float v = qh + k[(size_t)s * HEADS + h];
    float e = v > 0.f ? v : NEG_SLOPE * v;
    float w = __expf(e - mh) * inv;
    acc += w * (float)Hsrc[(size_t)s * HF + h * OUT_F + f];
  }
  red[t] = acc;
  __syncthreads();

  // mean over heads -> out[d, f]
  if (t < OUT_F) {
    float sumh = 0.f;
#pragma unroll
    for (int hh = 0; hh < HEADS; ++hh) sumh += red[hh * OUT_F + f];
    out[(size_t)d * OUT_F + f] = sumh * (1.0f / HEADS);
  }
}

// ---------------------------------------------------------------------------
extern "C" void kernel_launch(void* const* d_in, const int* in_sizes, int n_in,
                              void* d_out, int out_size, void* d_ws,
                              size_t ws_size, hipStream_t stream) {
  const float* x_src  = (const float*)d_in[0];
  const float* x_dst  = (const float*)d_in[1];
  const float* Wv     = (const float*)d_in[2];
  const float* bv     = (const float*)d_in[3];
  const float* Wq     = (const float*)d_in[4];
  const float* bq     = (const float*)d_in[5];
  const float* Wk     = (const float*)d_in[6];
  const float* bk     = (const float*)d_in[7];
  const int*   src_idx = (const int*)d_in[8];
  const int*   dst_idx = (const int*)d_in[9];

  const int n_src = in_sizes[0] / IN_F;
  const int n_dst = in_sizes[1] / IN_F;
  const int E     = in_sizes[8];
  float* out = (float*)d_out;

  // workspace carving (256B aligned)
  char* ws = (char*)d_ws;
  size_t off = 0;
  auto carve = [&](size_t bytes) {
    void* p = ws + off;
    off = (off + bytes + 255) & ~(size_t)255;
    return p;
  };
  _Float16* h_src = (_Float16*)carve((size_t)n_src * HF * sizeof(_Float16));
  _Float16* h_dst = (_Float16*)carve((size_t)n_dst * HF * sizeof(_Float16));
  float*    qarr  = (float*)carve((size_t)n_dst * HEADS * sizeof(float));
  float*    karr  = (float*)carve((size_t)n_src * HEADS * sizeof(float));
  _Float16* WT    = (_Float16*)carve((size_t)HF * IN_F * sizeof(_Float16));
  int*      seg   = (int*)carve((size_t)(n_dst + 1) * sizeof(int));
  (void)ws_size; (void)n_in; (void)out_size;

  // 1) weight transpose + f16 convert
  gat_wt_convert<<<(IN_F * HF + 255) / 256, 256, 0, stream>>>(Wv, WT);

  // 2) projections h = x @ Wv + bv (WMMA f16, f32 accumulate, f16 output)
  const int rtS = (n_src + 15) / 16;
  const int rtD = (n_dst + 15) / 16;
  gat_gemm256<<<(rtS + 7) / 8, 256, 0, stream>>>(x_src, WT, bv, h_src, n_src);
  gat_gemm256<<<(rtD + 7) / 8, 256, 0, stream>>>(x_dst, WT, bv, h_dst, n_dst);

  // 3) q = h_dst @ Wq + bq ; k = h_src @ Wk + bk
  gat_qk<<<(n_dst + 7) / 8, 256, 0, stream>>>(h_dst, Wq, bq, qarr, n_dst);
  gat_qk<<<(n_src + 7) / 8, 256, 0, stream>>>(h_src, Wk, bk, karr, n_src);

  // 4) segment offsets from sorted dst_idx
  gat_segstart<<<(n_dst + 256) / 256, 256, 0, stream>>>(dst_idx, E, n_dst, seg);

  // 5) fused edge softmax + aggregation + head mean
  gat_edge_aggr<<<n_dst, 256, 0, stream>>>(h_src, qarr, karr, src_idx, seg,
                                           out, n_dst);
}
